// ActiveBlockPromptBasis_66245575573738
// MI455X (gfx1250) — matrix operand order repaired and checked by reading, not compiled, not run on hardware
//
#include <hip/hip_runtime.h>
#include <hip/hip_bf16.h>

// MI455X / gfx1250, wave32. Heavy work routed through v_wmma_f32_16x16x32_f16.

typedef __attribute__((ext_vector_type(16))) _Float16 v16h;
typedef __attribute__((ext_vector_type(8)))  float    v8f;
typedef __attribute__((ext_vector_type(4)))  unsigned v4u;

union HFrag { v16h h; unsigned u[8]; v4u q[2]; };

__device__ __forceinline__ float gelu_exact(float v) {
    return 0.5f * v * (1.0f + erff(v * 0.70710678118654752f));
}

// ---------------------------------------------------------------------------
// Kernel 0: conv weights [co][ci][3][3] f32  ->  wT[co][tap*128+ci] f16
// (K contiguous per co so a B-fragment VGPR is one aligned b32 load)
// ---------------------------------------------------------------------------
__global__ __launch_bounds__(256)
void abpb_wconvert(const float* __restrict__ conv_w, _Float16* __restrict__ wT) {
    int i = blockIdx.x * 256 + threadIdx.x;
    if (i >= 128 * 1152) return;
    int co  = i / 1152;
    int rem = i - co * 1152;
    int tap = rem >> 7;     // 0..8
    int ci  = rem & 127;
    wT[i] = (_Float16)conv_w[(co * 128 + ci) * 9 + tap];
}

// ---------------------------------------------------------------------------
// Kernel 1: per-pixel routing MLPs + WMMA prompt mixing.
// Each thread owns one pixel; each wave then runs two 16x64x128 f16 GEMMs.
// Output: prompts NHWC f16  ws[pixel*128 + e]
// ---------------------------------------------------------------------------
__global__ __launch_bounds__(256)
void abpb_routing_mix(const float* __restrict__ x,
                      const float* __restrict__ flux,
                      const float* __restrict__ prompt,
                      const float* __restrict__ b1w, const float* __restrict__ b1b,
                      const float* __restrict__ b2w, const float* __restrict__ b2b,
                      const float* __restrict__ t1w, const float* __restrict__ t1b,
                      const float* __restrict__ t2w, const float* __restrict__ t2b,
                      _Float16* __restrict__ prompts_out)
{
    __shared__ float    s_b1w[8 * 64];
    __shared__ float    s_b1b[8];
    __shared__ float    s_b2w[8 * 8];
    __shared__ float    s_b2b[8];
    __shared__ float    s_t1w[6 * 128];
    __shared__ float    s_t1b[6];
    __shared__ float    s_t2w[6 * 6];
    __shared__ float    s_t2b[6];
    __shared__ _Float16 s_w48[256][64];      // per-pixel mixing weights, K padded 48->64
    __shared__ _Float16 s_promptT[128 * 64]; // [e][j] (j = t*8+k, padded to 64)

    const int tid = threadIdx.x;

    for (int i = tid; i < 8 * 64;  i += 256) s_b1w[i] = b1w[i];
    for (int i = tid; i < 6 * 128; i += 256) s_t1w[i] = t1w[i];
    if (tid < 8)  { s_b1b[tid] = b1b[tid]; s_b2b[tid] = b2b[tid]; }
    if (tid < 64)   s_b2w[tid] = b2w[tid];
    if (tid < 6)  { s_t1b[tid] = t1b[tid]; s_t2b[tid] = t2b[tid]; }
    if (tid < 36)   s_t2w[tid] = t2w[tid];
    for (int i = tid; i < 128 * 64; i += 256) {
        int j = i & 63;
        int e = i >> 6;
        // prompt is [6][8][128] f32 -> flat j*128+e
        s_promptT[i] = (j < 48) ? (_Float16)prompt[j * 128 + e] : (_Float16)0.0f;
    }
    __syncthreads();

    const long pix = (long)blockIdx.x * 256 + tid;   // [0, 8*256*256)
    const long b   = pix >> 16;
    const long hw  = pix & 65535;

    // ---- basis MLP: x[64] -> 8 -> gelu -> 8 -> softmax ----
    float h1[8];
    #pragma unroll
    for (int k = 0; k < 8; ++k) h1[k] = s_b1b[k];
    const float* xp = x + b * 64 * 65536 + hw;
    for (int c = 0; c < 64; ++c) {
        float xv = xp[(long)c * 65536];
        #pragma unroll
        for (int k = 0; k < 8; ++k) h1[k] += xv * s_b1w[k * 64 + c];
    }
    #pragma unroll
    for (int k = 0; k < 8; ++k) h1[k] = gelu_exact(h1[k]);
    float zb[8];
    float bmax = -1e30f;
    #pragma unroll
    for (int j = 0; j < 8; ++j) {
        float a = s_b2b[j];
        #pragma unroll
        for (int k = 0; k < 8; ++k) a += h1[k] * s_b2w[j * 8 + k];
        zb[j] = a; bmax = fmaxf(bmax, a);
    }
    float bsum = 0.f;
    #pragma unroll
    for (int j = 0; j < 8; ++j) { zb[j] = __expf(zb[j] - bmax); bsum += zb[j]; }
    float binv = 1.0f / bsum;
    #pragma unroll
    for (int j = 0; j < 8; ++j) zb[j] *= binv;

    // ---- task MLP: flux[128] (nan->0) -> 6 -> gelu -> 6 -> softmax ----
    float t1[6];
    #pragma unroll
    for (int k = 0; k < 6; ++k) t1[k] = s_t1b[k];
    const float* fp = flux + b * 128 * 65536 + hw;
    for (int c = 0; c < 128; ++c) {
        float fv = fp[(long)c * 65536];
        if (fv != fv) fv = 0.0f;   // NaN -> 0
        #pragma unroll
        for (int k = 0; k < 6; ++k) t1[k] += fv * s_t1w[k * 128 + c];
    }
    #pragma unroll
    for (int k = 0; k < 6; ++k) t1[k] = gelu_exact(t1[k]);
    float zt[6];
    float tmax = -1e30f;
    #pragma unroll
    for (int j = 0; j < 6; ++j) {
        float a = s_t2b[j];
        #pragma unroll
        for (int k = 0; k < 6; ++k) a += t1[k] * s_t2w[j * 6 + k];
        zt[j] = a; tmax = fmaxf(tmax, a);
    }
    float tsum = 0.f;
    #pragma unroll
    for (int j = 0; j < 6; ++j) { zt[j] = __expf(zt[j] - tmax); tsum += zt[j]; }
    float tinv = 1.0f / tsum;
    #pragma unroll
    for (int j = 0; j < 6; ++j) zt[j] *= tinv;

    // ---- outer product -> 48 mixing weights into LDS (f16, padded to 64) ----
    #pragma unroll
    for (int t = 0; t < 6; ++t)
        #pragma unroll
        for (int k = 0; k < 8; ++k)
            s_w48[tid][t * 8 + k] = (_Float16)(zt[t] * zb[k]);
    #pragma unroll
    for (int j = 48; j < 64; ++j) s_w48[tid][j] = (_Float16)0.0f;
    __syncthreads();

    // ---- WMMA mixing: per wave, 2 tiles of (16 px) x (K=64) x (N=128) ----
    const int wave  = tid >> 5;
    const int lane  = tid & 31;
    const int laneM = lane & 15;
    const int hi    = lane >> 4;     // lane half (0: lanes 0-15, 1: lanes 16-31)

    #pragma unroll
    for (int m = 0; m < 2; ++m) {
        const int rowBase = wave * 32 + m * 16;
        const int row     = rowBase + laneM;
        HFrag a0, a1;
        #pragma unroll
        for (int v = 0; v < 8; ++v) {
            int koff = ((v < 4) ? (2 * v) : (16 + 2 * (v - 4))) + hi * 8;
            a0.u[v] = *(const unsigned*)&s_w48[row][koff];
            a1.u[v] = *(const unsigned*)&s_w48[row][koff + 32];
        }
        const long pbase = (long)blockIdx.x * 256 + rowBase;
        for (int n0 = 0; n0 < 8; ++n0) {
            const int e = n0 * 16 + laneM;
            HFrag bf0, bf1;
            #pragma unroll
            for (int v = 0; v < 8; ++v) {
                int j = ((v < 4) ? (2 * v) : (16 + 2 * (v - 4))) + hi * 8;
                bf0.u[v] = *(const unsigned*)&s_promptT[e * 64 + j];
                bf1.u[v] = *(const unsigned*)&s_promptT[e * 64 + j + 32];
            }
            v8f acc = {0.f, 0.f, 0.f, 0.f, 0.f, 0.f, 0.f, 0.f};
            acc = __builtin_amdgcn_wmma_f32_16x16x32_f16(false, a0.h, false, bf0.h,
                                                         (short)0, acc, false, false);
            acc = __builtin_amdgcn_wmma_f32_16x16x32_f16(false, a1.h, false, bf1.h,
                                                         (short)0, acc, false, false);
            #pragma unroll
            for (int r = 0; r < 8; ++r) {
                long pxi = pbase + r + hi * 8;
                prompts_out[pxi * 128 + e] = (_Float16)acc[r];
            }
        }
    }
}

// ---------------------------------------------------------------------------
// Kernel 2: 3x3 conv 128->128 as implicit GEMM with f16 WMMA, f32 accumulate.
// Block = 8 waves = 256 output pixels (one full (b,y) row); each wave owns two
// 16-pixel M-tiles so every staged B-fragment feeds two WMMAs.
// K = 9 taps * 128 ci, stepped 32 at a time; weight slice staged in LDS.
// ---------------------------------------------------------------------------
__global__ __launch_bounds__(256)
void abpb_conv3x3(const _Float16* __restrict__ prompts,
                  const _Float16* __restrict__ wT,
                  float* __restrict__ out)
{
    __shared__ _Float16 s_B[128 * 32];       // weight slice [co][k], 8 KB
    __shared__ float    s_out[8 * 16 * 128]; // 64 KB per-wave transpose staging

    const int tid   = threadIdx.x;
    const int wave  = tid >> 5;
    const int lane  = tid & 31;
    const int laneM = lane & 15;
    const int hi    = lane >> 4;

    const int y = blockIdx.x;
    const int b = blockIdx.y;

    v8f acc[2][8];
    #pragma unroll
    for (int m = 0; m < 2; ++m)
        #pragma unroll
        for (int n = 0; n < 8; ++n)
            acc[m][n] = (v8f){0.f, 0.f, 0.f, 0.f, 0.f, 0.f, 0.f, 0.f};

    const _Float16* pb = prompts + (long)b * 65536 * 128;

    for (int kb = 0; kb < 36; ++kb) {
        const int tap = kb >> 2;
        const int cib = (kb & 3) * 32;
        const int dy  = tap / 3;
        const int dx  = tap - dy * 3;

        __syncthreads();
        for (int i = tid; i < 4096; i += 256)      // stage 32x128 weight slice
            s_B[i] = wT[(i >> 5) * 1152 + kb * 32 + (i & 31)];
        if (kb + 1 < 36) {                         // hint next slice into cache
            int pi = tid * 16;                     // covers all 128 co rows
            __builtin_prefetch(&wT[(pi >> 5) * 1152 + (kb + 1) * 32 + (pi & 31)], 0, 3);
        }
        __syncthreads();

        const int yy = y + dy - 1;
        HFrag a[2];
        #pragma unroll
        for (int m = 0; m < 2; ++m) {
            const int xx = wave * 32 + m * 16 + laneM + dx - 1;
            const bool valid = (yy >= 0) && (yy < 256) && (xx >= 0) && (xx < 256);
            const _Float16* ap = pb + ((long)yy * 256 + xx) * 128 + cib + hi * 8;
            if (valid) {   // two aligned 16B chunks: bytes [hi*16,+16) and [32+hi*16,+16)
                a[m].q[0] = *(const v4u*)(ap);
                a[m].q[1] = *(const v4u*)(ap + 16);
            } else {
                a[m].q[0] = (v4u){0u, 0u, 0u, 0u};
                a[m].q[1] = (v4u){0u, 0u, 0u, 0u};
            }
        }

        #pragma unroll
        for (int n0 = 0; n0 < 8; ++n0) {
            const int co = n0 * 16 + laneM;
            HFrag bf;
            #pragma unroll
            for (int v = 0; v < 8; ++v)
                bf.u[v] = *(const unsigned*)&s_B[co * 32 + ((v < 4) ? (2 * v) : (16 + 2 * (v - 4))) + hi * 8];
            acc[0][n0] = __builtin_amdgcn_wmma_f32_16x16x32_f16(false, a[0].h, false, bf.h,
                                                                (short)0, acc[0][n0], false, false);
            acc[1][n0] = __builtin_amdgcn_wmma_f32_16x16x32_f16(false, a[1].h, false, bf.h,
                                                                (short)0, acc[1][n0], false, false);
        }
    }

    // Transpose each M-tile through per-wave-private LDS (DS ops are in-order
    // per wave, so reusing the region across m needs no barrier), then store
    // NCHW with 64B-contiguous segments per lane-half.
    float* so = s_out + wave * 16 * 128;
    #pragma unroll
    for (int m = 0; m < 2; ++m) {
        #pragma unroll
        for (int n0 = 0; n0 < 8; ++n0)
            #pragma unroll
            for (int r = 0; r < 8; ++r)
                so[(r + hi * 8) * 128 + n0 * 16 + laneM] = acc[m][n0][r];
        float* ob = out + (long)b * 128 * 65536 + (long)y * 256 + wave * 32 + m * 16;
        for (int cc = 0; cc < 128; cc += 2) {
            int co = cc + hi;
            ob[(long)co * 65536 + laneM] = so[laneM * 128 + co];
        }
    }
}

// ---------------------------------------------------------------------------
extern "C" void kernel_launch(void* const* d_in, const int* in_sizes, int n_in,
                              void* d_out, int out_size, void* d_ws, size_t ws_size,
                              hipStream_t stream) {
    const float* x      = (const float*)d_in[0];
    const float* flux   = (const float*)d_in[1];
    const float* prompt = (const float*)d_in[2];
    const float* conv_w = (const float*)d_in[3];
    const float* b1w    = (const float*)d_in[4];
    const float* b1b    = (const float*)d_in[5];
    const float* b2w    = (const float*)d_in[6];
    const float* b2b    = (const float*)d_in[7];
    const float* t1w    = (const float*)d_in[8];
    const float* t1b    = (const float*)d_in[9];
    const float* t2w    = (const float*)d_in[10];
    const float* t2b    = (const float*)d_in[11];
    float* out = (float*)d_out;

    // Workspace layout: prompts f16 NHWC (134 217 728 B) | wT f16 (294 912 B)
    _Float16* prompts = (_Float16*)d_ws;
    _Float16* wT      = (_Float16*)((char*)d_ws + (size_t)8 * 65536 * 128 * 2);

    abpb_wconvert<<<dim3((128 * 1152 + 255) / 256), dim3(256), 0, stream>>>(conv_w, wT);

    abpb_routing_mix<<<dim3(2048), dim3(256), 0, stream>>>(
        x, flux, prompt, b1w, b1b, b2w, b2b, t1w, t1b, t2w, t2b, prompts);

    abpb_conv3x3<<<dim3(256, 8), dim3(256), 0, stream>>>(prompts, wT, out);
}
